// Decoder_12240656794363
// MI455X (gfx1250) — matrix operand order repaired and checked by reading, not compiled, not run on hardware
//
#include <hip/hip_runtime.h>
#include <hip/hip_bf16.h>
#include <math.h>

// ---------------------------------------------------------------------------
// HyperLSTM + MDN for MI455X (gfx1250, wave32, WMMA bf16 16x16x32).
// - Sequential scan: ONE persistent kernel (32 WGs) with grid barriers.
// - All GEMMs: v_wmma_f32_16x16x32_bf16, weights pre-packed in B-fragment
//   order (one contiguous v16bf load per lane per k-tile, L2-resident).
// - Per-step activations (h / x_t / hh / z3) staged into LDS with
//   GLOBAL_LOAD_ASYNC_TO_LDS_B128 (+ s_wait_asynccnt), WMMA A-fragments then
//   come from ds_load instead of redundant global reads.
// ---------------------------------------------------------------------------

#define NWG    32
#define TPB    256
#define NWAVES (NWG * (TPB / 32))

typedef __bf16 bf16_t;
typedef bf16_t v16bf __attribute__((ext_vector_type(16)));
typedef bf16_t v8bf  __attribute__((ext_vector_type(8)));
typedef float  v8f   __attribute__((ext_vector_type(8)));

#define MIX    20
#define LATENT 128
#define IN_DIM 133
#define IN_PAD 160
#define Hh     1024
#define HYd    256
#define NSEQ   128
#define Bb     64
#define OUTD   123
#define OUTP   128

// LDS layout (bytes) inside the scan kernel
#define LDS_H_OFF   0                       // h: 64x1024 bf16 = 131072 B (aliased by z3 in phase 4)
#define LDS_X_OFF   131072                  // x_t: 64x160 bf16 = 20480 B
#define LDS_HH_OFF  151552                  // hh: 64x256 bf16 = 32768 B
#define LDS_BYTES   184320

// ---- workspace layout (bytes) ---------------------------------------------
static constexpr size_t algn(size_t x) { return (x + 255) & ~(size_t)255; }

static constexpr size_t OFF_BAR   = 0;                      // 2 u32 barrier
static constexpr size_t OFF_WH    = 256;                                   static constexpr size_t SZ_WH    = (size_t)1024*4096*2;
static constexpr size_t OFF_WX    = algn(OFF_WH   + SZ_WH);                static constexpr size_t SZ_WX    = (size_t)160*4096*2;
static constexpr size_t OFF_WXHX  = algn(OFF_WX   + SZ_WX);                static constexpr size_t SZ_WXHX  = (size_t)160*1024*2;
static constexpr size_t OFF_WXHH  = algn(OFF_WXHX + SZ_WXHX);              static constexpr size_t SZ_WXHH  = (size_t)1024*1024*2;
static constexpr size_t OFF_WHH   = algn(OFF_WXHH + SZ_WXHH);              static constexpr size_t SZ_WHH   = (size_t)256*1024*2;
static constexpr size_t OFF_WZ    = algn(OFF_WHH  + SZ_WHH);               static constexpr size_t SZ_WZ    = (size_t)3*256*256*2;
static constexpr size_t OFF_DPK   = algn(OFF_WZ   + SZ_WZ);                static constexpr size_t SZ_DPK   = (size_t)12*64*1024*2;
static constexpr size_t OFF_PRJ   = algn(OFF_DPK  + SZ_DPK);               static constexpr size_t SZ_PRJ   = (size_t)1024*128*2;
static constexpr size_t OFF_FCIN  = algn(OFF_PRJ  + SZ_PRJ);               static constexpr size_t SZ_FCIN  = (size_t)128*2560*2;
static constexpr size_t OFF_XBF   = algn(OFF_FCIN + SZ_FCIN);              static constexpr size_t SZ_XBF   = (size_t)NSEQ*Bb*IN_PAD*2;
static constexpr size_t OFF_ZBF   = algn(OFF_XBF  + SZ_XBF);               static constexpr size_t SZ_ZBF   = (size_t)Bb*LATENT*2;
static constexpr size_t OFF_HBF   = algn(OFF_ZBF  + SZ_ZBF);               static constexpr size_t SZ_HBF   = (size_t)Bb*Hh*2;
static constexpr size_t OFF_HHBF  = algn(OFF_HBF  + SZ_HBF);               static constexpr size_t SZ_HHBF  = (size_t)Bb*HYd*2;
static constexpr size_t OFF_Z3    = algn(OFF_HHBF + SZ_HHBF);              static constexpr size_t SZ_Z3    = (size_t)3*Bb*HYd*2;
static constexpr size_t OFF_OUTS  = algn(OFF_Z3   + SZ_Z3);                static constexpr size_t SZ_OUTS  = (size_t)NSEQ*Bb*Hh*2;
static constexpr size_t OFF_CBUF  = algn(OFF_OUTS + SZ_OUTS);              static constexpr size_t SZ_CBUF  = (size_t)Bb*Hh*4;
static constexpr size_t OFF_CHBUF = algn(OFF_CBUF + SZ_CBUF);              static constexpr size_t SZ_CHBUF = (size_t)Bb*HYd*4;
static constexpr size_t OFF_GBUF  = algn(OFF_CHBUF+ SZ_CHBUF);             static constexpr size_t SZ_GBUF  = (size_t)Bb*1024*4;
static constexpr size_t OFF_HWH   = algn(OFF_GBUF + SZ_GBUF);              static constexpr size_t SZ_HWH   = (size_t)Bb*4096*4;
static constexpr size_t OFF_XWX   = algn(OFF_HWH  + SZ_HWH);               static constexpr size_t SZ_XWX   = (size_t)Bb*4096*4;
static constexpr size_t OFF_PRE   = algn(OFF_XWX  + SZ_XWX);               static constexpr size_t SZ_PRE   = (size_t)Bb*4096*4;
static constexpr size_t OFF_DEC   = algn(OFF_PRE  + SZ_PRE);               static constexpr size_t SZ_DEC   = (size_t)NSEQ*Bb*OUTP*4;

// ---- device helpers --------------------------------------------------------
__device__ __forceinline__ float sigm(float x) { return 1.0f / (1.0f + expf(-x)); }

// 16x16 C-tile GEMM accumulate: A row-major bf16 (already offset to tile row,
// may live in LDS -> ds_load after addrspace inference), B pre-packed
// fragments (512 bf16 per 32x16 k-tile, lane-contiguous 32-byte load).
__device__ __forceinline__ v8f wmma_acc(const bf16_t* __restrict__ A, int lda,
                                        const bf16_t* __restrict__ Bpk, int ktiles,
                                        v8f acc) {
  const int lane = threadIdx.x & 31;
  const bf16_t* arow = A + (size_t)(lane & 15) * lda + ((lane >= 16) ? 8 : 0);
  const bf16_t* bp   = Bpk + (size_t)lane * 16;
  for (int kt = 0; kt < ktiles; ++kt) {
    v8bf alo = *(const v8bf*)(arow);
    v8bf ahi = *(const v8bf*)(arow + 16);
    v16bf a;
#pragma unroll
    for (int j = 0; j < 8; ++j) { a[j] = alo[j]; a[j + 8] = ahi[j]; }
    v16bf b = *(const v16bf*)(bp);
    acc = __builtin_amdgcn_wmma_f32_16x16x32_bf16(false, a, false, b,
                                                  (short)0, acc, false, false);
    arow += 32;
    bp   += 512;
  }
  return acc;
}

__device__ __forceinline__ v8f vzero8() { v8f a;
#pragma unroll
  for (int r = 0; r < 8; ++r) a[r] = 0.0f; return a; }

__device__ __forceinline__ void grid_barrier(unsigned* bar, int nwg) {
  __threadfence();
  __syncthreads();
  if (threadIdx.x == 0) {
    unsigned* cnt = bar;
    unsigned* gen = bar + 1;
    unsigned g = __atomic_load_n(gen, __ATOMIC_ACQUIRE);
    unsigned prev = __atomic_fetch_add(cnt, 1u, __ATOMIC_ACQ_REL);
    if (prev == (unsigned)nwg - 1u) {
      __atomic_store_n(cnt, 0u, __ATOMIC_RELAXED);
      __atomic_fetch_add(gen, 1u, __ATOMIC_ACQ_REL);
    } else {
      while (__atomic_load_n(gen, __ATOMIC_ACQUIRE) == g)
        __builtin_amdgcn_s_sleep(2);
    }
    __threadfence();
  }
  __syncthreads();
}

// Block-cooperative async copy global -> LDS (16 B per lane per issue).
// `bytes` must be a multiple of TPB*16 (4096) so control flow is uniform.
// GVS addressing: SGPR64 global base + VGPR 32-bit offset; VDST = LDS byte
// offset (low 32 bits of the flat LDS pointer). Tracked by ASYNCcnt.
__device__ __forceinline__ void stage_async(const bf16_t* __restrict__ gsrc,
                                            bf16_t* __restrict__ ldst,
                                            unsigned bytes) {
  unsigned lds_base = (unsigned)(size_t)(void*)ldst;   // LDS aperture: addr[31:0] = LDS offset
  unsigned long long gbase = (unsigned long long)(size_t)gsrc;
  for (unsigned off = threadIdx.x * 16u; off < bytes; off += TPB * 16u) {
    unsigned laddr = lds_base + off;
    asm volatile("global_load_async_to_lds_b128 %0, %1, %2"
                 :: "v"(laddr), "v"(off), "s"(gbase) : "memory");
  }
  asm volatile("s_wait_asynccnt 0x0" ::: "memory");
}

// ---- packing / conversion kernels -----------------------------------------
// Pack a KxN f32 weight into WMMA-B fragment order (Kpad x Npad, bf16).
__global__ void pack_b(const float* __restrict__ src, int srcLd,
                       bf16_t* __restrict__ dst, int K, int N, int Kpad, int Npad) {
  long idx = (long)blockIdx.x * blockDim.x + threadIdx.x;
  long total = (long)Kpad * Npad;
  if (idx >= total) return;
  int ktiles = Kpad >> 5;
  long tile  = idx >> 9;
  int within = (int)(idx & 511);
  int lane   = within >> 4;
  int i      = within & 15;
  int ktile  = (int)(tile % ktiles);
  int ntile  = (int)(tile / ktiles);
  int n  = ntile * 16 + (lane & 15);
  int kk = ((i < 8) ? i : (i + 8)) + ((lane >= 16) ? 8 : 0);   // B 32x16: lanes 0-15 K 0..7,16..23; 16-31 K 8..15,24..31
  int k  = ktile * 32 + kk;
  float v = (k < K && n < N) ? src[(long)k * srcLd + n] : 0.0f;
  dst[idx] = (bf16_t)v;
}

__global__ void cvt_rowpad(const float* __restrict__ src, bf16_t* __restrict__ dst,
                           long rows, int cin, int cpad) {
  long idx = (long)blockIdx.x * blockDim.x + threadIdx.x;
  if (idx >= rows * (long)cpad) return;
  long r = idx / cpad;
  int  c = (int)(idx - r * cpad);
  dst[idx] = (c < cin) ? (bf16_t)src[r * (long)cin + c] : (bf16_t)0.0f;
}

// ---- initial state: s0 = tanh(z @ fc_in_w + b); scatter into h0/c0/hh0/ch0
__global__ __launch_bounds__(TPB) void init_state(const bf16_t* __restrict__ zbf,
                                                  const bf16_t* __restrict__ fcin_pk,
                                                  const float* __restrict__ fcin_b,
                                                  bf16_t* __restrict__ h_bf,
                                                  float* __restrict__ cbuf,
                                                  bf16_t* __restrict__ hh_bf,
                                                  float* __restrict__ chbuf) {
  int wave = (blockIdx.x * TPB + threadIdx.x) >> 5;
  if (wave >= 640) return;                    // 4 m-tiles x 160 n-tiles
  int lane = threadIdx.x & 31;
  int mtile = wave & 3, ntile = wave >> 2;
  v8f acc = vzero8();
  acc = wmma_acc(zbf + (size_t)(mtile * 16) * LATENT, LATENT,
                 fcin_pk + (size_t)ntile * 4 * 512, 4, acc);
  int col = ntile * 16 + (lane & 15);
  float bias = fcin_b[col];
  int mb = (lane >= 16) ? 8 : 0;
#pragma unroll
  for (int r = 0; r < 8; ++r) {
    int m = mtile * 16 + r + mb;
    float v = tanhf(acc[r] + bias);
    if (col < 1024)       h_bf[(size_t)m * 1024 + col]          = (bf16_t)v;
    else if (col < 2048)  cbuf[(size_t)m * 1024 + (col - 1024)] = v;
    else if (col < 2304)  hh_bf[(size_t)m * 256 + (col - 2048)] = (bf16_t)v;
    else                  chbuf[(size_t)m * 256 + (col - 2304)] = v;
  }
}

// ---- the sequential scan: one persistent kernel, 5 phases per timestep ----
__global__ __launch_bounds__(TPB) void scan_kernel(char* __restrict__ ws,
                                                   const float* __restrict__ b_hy,
                                                   const float* __restrict__ bzx,
                                                   const float* __restrict__ bzh,
                                                   const float* __restrict__ b0) {
  extern __shared__ __align__(16) char smem[];
  bf16_t* sh_h  = (bf16_t*)(smem + LDS_H_OFF);    // h (phase 1) / z3 (phase 4)
  bf16_t* sh_x  = (bf16_t*)(smem + LDS_X_OFF);
  bf16_t* sh_hh = (bf16_t*)(smem + LDS_HH_OFF);

  unsigned*     bar     = (unsigned*)(ws + OFF_BAR);
  const bf16_t* Wh_pk   = (const bf16_t*)(ws + OFF_WH);
  const bf16_t* Wx_pk   = (const bf16_t*)(ws + OFF_WX);
  const bf16_t* WxhX_pk = (const bf16_t*)(ws + OFF_WXHX);
  const bf16_t* WxhH_pk = (const bf16_t*)(ws + OFF_WXHH);
  const bf16_t* Whh_pk  = (const bf16_t*)(ws + OFF_WHH);
  const bf16_t* Wz_pk   = (const bf16_t*)(ws + OFF_WZ);
  const bf16_t* D_pk    = (const bf16_t*)(ws + OFF_DPK);
  const bf16_t* xbf     = (const bf16_t*)(ws + OFF_XBF);
  bf16_t*       h_bf    = (bf16_t*)(ws + OFF_HBF);
  bf16_t*       hh_bf   = (bf16_t*)(ws + OFF_HHBF);
  bf16_t*       z3      = (bf16_t*)(ws + OFF_Z3);
  bf16_t*       outs_bf = (bf16_t*)(ws + OFF_OUTS);
  float*        cbuf    = (float*)(ws + OFF_CBUF);
  float*        chbuf   = (float*)(ws + OFF_CHBUF);
  float*        gbuf    = (float*)(ws + OFF_GBUF);
  float*        hwh     = (float*)(ws + OFF_HWH);
  float*        xwx     = (float*)(ws + OFF_XWX);
  float*        pre     = (float*)(ws + OFF_PRE);

  const int tid  = blockIdx.x * TPB + threadIdx.x;
  const int wave = tid >> 5;
  const int lane = threadIdx.x & 31;
  const int nlo  = lane & 15;
  const int mb   = (lane >= 16) ? 8 : 0;

  for (int t = 0; t < NSEQ; ++t) {
    const bf16_t* xb = xbf + (size_t)t * Bb * IN_PAD;

    // stage A-operands for phase 1 into LDS (async DMA, ASYNCcnt-tracked)
    stage_async(h_bf,  sh_h,  (unsigned)(Bb * Hh * 2));      // 128 KB
    stage_async(xb,    sh_x,  (unsigned)(Bb * IN_PAD * 2));  //  20 KB
    stage_async(hh_bf, sh_hh, (unsigned)(Bb * HYd * 2));     //  32 KB
    __syncthreads();

    // phase 1: g = x@WxhX + h@WxhH + hh@Whh + b_hy ; hWh = h@Wh ; xWx = x@Wx
    for (int tile = wave; tile < 2304; tile += NWAVES) {
      if (tile < 256) {
        int mtile = tile & 3, ntile = tile >> 2;
        v8f acc = vzero8();
        acc = wmma_acc(sh_x  + (size_t)(mtile * 16) * IN_PAD, IN_PAD,
                       WxhX_pk + (size_t)ntile * 5 * 512, 5, acc);
        acc = wmma_acc(sh_h  + (size_t)(mtile * 16) * Hh, Hh,
                       WxhH_pk + (size_t)ntile * 32 * 512, 32, acc);
        acc = wmma_acc(sh_hh + (size_t)(mtile * 16) * HYd, HYd,
                       Whh_pk + (size_t)ntile * 8 * 512, 8, acc);
        int col = ntile * 16 + nlo;
        float bias = b_hy[col];
        float* dst = gbuf + (size_t)(mtile * 16) * 1024 + col;
#pragma unroll
        for (int r = 0; r < 8; ++r) dst[(size_t)(r + mb) * 1024] = acc[r] + bias;
      } else if (tile < 1280) {
        int u = tile - 256, mtile = u & 3, ntile = u >> 2;
        v8f acc = vzero8();
        acc = wmma_acc(sh_h + (size_t)(mtile * 16) * Hh, Hh,
                       Wh_pk + (size_t)ntile * 32 * 512, 32, acc);
        float* dst = hwh + (size_t)(mtile * 16) * 4096 + ntile * 16 + nlo;
#pragma unroll
        for (int r = 0; r < 8; ++r) dst[(size_t)(r + mb) * 4096] = acc[r];
      } else {
        int u = tile - 1280, mtile = u & 3, ntile = u >> 2;
        v8f acc = vzero8();
        acc = wmma_acc(sh_x + (size_t)(mtile * 16) * IN_PAD, IN_PAD,
                       Wx_pk + (size_t)ntile * 5 * 512, 5, acc);
        float* dst = xwx + (size_t)(mtile * 16) * 4096 + ntile * 16 + nlo;
#pragma unroll
        for (int r = 0; r < 8; ++r) dst[(size_t)(r + mb) * 4096] = acc[r];
      }
    }
    grid_barrier(bar, NWG);

    // phase 2: hyper cell -> ch, hh
    for (int idx = tid; idx < Bb * HYd; idx += NWG * TPB) {
      int m = idx >> 8, n = idx & 255;
      const float* gr = gbuf + (size_t)m * 1024;
      float hi = gr[n], hf = gr[256 + n], hg = gr[512 + n], ho = gr[768 + n];
      float chv = sigm(hf) * chbuf[idx] + sigm(hi) * tanhf(hg);
      chbuf[idx] = chv;
      hh_bf[idx] = (bf16_t)(sigm(ho) * tanhf(chv));
    }
    grid_barrier(bar, NWG);

    // stage updated hh for phase 3
    stage_async(hh_bf, sh_hh, (unsigned)(Bb * HYd * 2));
    __syncthreads();

    // phase 3: zx/zh/zb = hh @ Wz* (+bias) -> bf16
    if (wave < 192) {
      int mat = wave / 64, rem = wave % 64;
      int mtile = rem & 3, ntile = rem >> 2;
      v8f acc = vzero8();
      acc = wmma_acc(sh_hh + (size_t)(mtile * 16) * HYd, HYd,
                     Wz_pk + (size_t)mat * 65536 + (size_t)ntile * 8 * 512, 8, acc);
      int col = ntile * 16 + nlo;
      float bias = (mat == 0) ? bzx[col] : (mat == 1) ? bzh[col] : 0.0f;
      bf16_t* dst = z3 + (size_t)mat * Bb * HYd + (size_t)(mtile * 16) * HYd + col;
#pragma unroll
      for (int r = 0; r < 8; ++r) dst[(size_t)(r + mb) * HYd] = (bf16_t)(acc[r] + bias);
    }
    grid_barrier(bar, NWG);

    // stage z3 into the (now dead) h region for phase 4
    stage_async(z3, sh_h, (unsigned)(3 * Bb * HYd * 2));     // 96 KB
    __syncthreads();
    bf16_t* sh_z3 = sh_h;

    // phase 4: pre = sx*xWx + sh*hWh + sb + b0  (s* = z*-slice @ D*[gate])
    for (int tile = wave; tile < 1024; tile += NWAVES) {
      int mtile = tile & 3, rest = tile >> 2;
      int gate = rest & 3, ntile = rest >> 2;          // ntile in [0,64)
      size_t aoff = (size_t)(mtile * 16) * HYd + gate * 64;
      v8f sx = vzero8(), shv = vzero8(), sb = vzero8();
      sx  = wmma_acc(sh_z3 + 0 * Bb * HYd + aoff, HYd,
                     D_pk + ((size_t)(0 * 4 + gate)) * 65536 + (size_t)ntile * 2 * 512, 2, sx);
      shv = wmma_acc(sh_z3 + 1 * Bb * HYd + aoff, HYd,
                     D_pk + ((size_t)(1 * 4 + gate)) * 65536 + (size_t)ntile * 2 * 512, 2, shv);
      sb  = wmma_acc(sh_z3 + 2 * Bb * HYd + aoff, HYd,
                     D_pk + ((size_t)(2 * 4 + gate)) * 65536 + (size_t)ntile * 2 * 512, 2, sb);
      int ncol = ntile * 16 + nlo;
      int col  = gate * 1024 + ncol;
      float b0v = b0[gate * 1024 + ncol];
#pragma unroll
      for (int r = 0; r < 8; ++r) {
        size_t row = (size_t)(mtile * 16 + r + mb);
        float pv = sx[r] * xwx[row * 4096 + col] + shv[r] * hwh[row * 4096 + col] + sb[r] + b0v;
        pre[row * 4096 + col] = pv;
      }
    }
    grid_barrier(bar, NWG);

    // phase 5: main LSTM cell -> c, h (store h as bf16 for next A and outs)
    for (int idx = tid; idx < Bb * Hh; idx += NWG * TPB) {
      int m = idx >> 10, n = idx & 1023;
      const float* pr = pre + (size_t)m * 4096;
      float mi = pr[n], mf = pr[1024 + n], mg = pr[2048 + n], mo = pr[3072 + n];
      float cv = sigm(mf) * cbuf[idx] + sigm(mi) * tanhf(mg);
      cbuf[idx] = cv;
      bf16_t hb = (bf16_t)(sigm(mo) * tanhf(cv));
      h_bf[idx] = hb;
      outs_bf[(size_t)t * (Bb * Hh) + idx] = hb;
    }
    grid_barrier(bar, NWG);
  }
}

// ---- time-parallel projection: dec = outs @ fc_proj_w + b -----------------
__global__ __launch_bounds__(TPB) void proj_kernel(const bf16_t* __restrict__ outs,
                                                   const bf16_t* __restrict__ pk,
                                                   const float* __restrict__ pb,
                                                   float* __restrict__ dec) {
  int wave = (blockIdx.x * TPB + threadIdx.x) >> 5;
  if (wave >= 4096) return;                   // 512 m-tiles x 8 n-tiles
  int lane = threadIdx.x & 31;
  int ntile = wave & 7, mtile = wave >> 3;
  v8f acc = vzero8();
  acc = wmma_acc(outs + (size_t)(mtile * 16) * Hh, Hh,
                 pk + (size_t)ntile * 32 * 512, 32, acc);
  int col = ntile * 16 + (lane & 15);
  float bias = (col < OUTD) ? pb[col] : 0.0f;
  int mb = (lane >= 16) ? 8 : 0;
#pragma unroll
  for (int r = 0; r < 8; ++r)
    dec[(size_t)(mtile * 16 + r + mb) * OUTP + col] = acc[r] + bias;
}

// ---- MDN head. NOTE: reference's mw softmax is over a SINGLETON axis -> 1.0
__global__ __launch_bounds__(TPB) void mdn_kernel(const float* __restrict__ dec,
                                                  float* __restrict__ out) {
  int tb = blockIdx.x * TPB + threadIdx.x;
  if (tb >= NSEQ * Bb) return;
  int t = tb >> 6, b = tb & 63;
  const float* row = dec + (size_t)tb * OUTP;
  const size_t blk = (size_t)NSEQ * MIX * Bb;           // 163840
#pragma unroll 4
  for (int m = 0; m < MIX; ++m) {
    size_t o = ((size_t)t * MIX + m) * Bb + b;
    out[0 * blk + o] = 1.0f;                            // softmax(axis=3) on size-1 axis
    out[1 * blk + o] = row[6 * m + 1];
    out[2 * blk + o] = row[6 * m + 2];
    out[3 * blk + o] = expf(row[6 * m + 3]);            // SQRT_T == 1
    out[4 * blk + o] = expf(row[6 * m + 4]);
    out[5 * blk + o] = tanhf(row[6 * m + 5]);
  }
  float p0 = row[120], p1 = row[121], p2 = row[122];
  float mx = fmaxf(p0, fmaxf(p1, p2));
  float e0 = expf(p0 - mx), e1 = expf(p1 - mx), e2 = expf(p2 - mx);
  float s = e0 + e1 + e2;
  float* pen = out + 6 * blk + (size_t)tb * 3;
  pen[0] = e0 / s; pen[1] = e1 / s; pen[2] = e2 / s;
}

// ---- host side -------------------------------------------------------------
static inline void launch_pack(const float* src, int srcLd, bf16_t* dst,
                               int K, int N, int Kpad, int Npad, hipStream_t s) {
  long total = (long)Kpad * Npad;
  int blocks = (int)((total + TPB - 1) / TPB);
  pack_b<<<blocks, TPB, 0, s>>>(src, srcLd, dst, K, N, Kpad, Npad);
}

extern "C" void kernel_launch(void* const* d_in, const int* in_sizes, int n_in,
                              void* d_out, int out_size, void* d_ws, size_t ws_size,
                              hipStream_t stream) {
  (void)in_sizes; (void)n_in; (void)out_size; (void)ws_size;
  char* ws = (char*)d_ws;
  const float* z        = (const float*)d_in[0];
  const float* strokes  = (const float*)d_in[1];
  const float* fc_in_w  = (const float*)d_in[2];
  const float* fc_in_b  = (const float*)d_in[3];
  const float* fc_prj_w = (const float*)d_in[4];
  const float* fc_prj_b = (const float*)d_in[5];
  const float* Wx       = (const float*)d_in[6];
  const float* Wh       = (const float*)d_in[7];
  const float* b0       = (const float*)d_in[8];
  const float* Wxh_hy   = (const float*)d_in[9];
  const float* Whh_hy   = (const float*)d_in[10];
  const float* b_hy     = (const float*)d_in[11];
  const float* Wzx      = (const float*)d_in[12];
  const float* bzx      = (const float*)d_in[13];
  const float* Wzh      = (const float*)d_in[14];
  const float* bzh      = (const float*)d_in[15];
  const float* Wzb      = (const float*)d_in[16];
  const float* Dx       = (const float*)d_in[17];
  const float* Dh       = (const float*)d_in[18];
  const float* Db       = (const float*)d_in[19];

  // zero the grid-barrier counters (graph-capture-safe)
  hipMemsetAsync(ws + OFF_BAR, 0, 256, stream);

  // pack weights into WMMA B-fragment order (bf16)
  launch_pack(Wh,              4096, (bf16_t*)(ws + OFF_WH),   1024, 4096, 1024, 4096, stream);
  launch_pack(Wx,              4096, (bf16_t*)(ws + OFF_WX),    133, 4096,  160, 4096, stream);
  launch_pack(Wxh_hy,          1024, (bf16_t*)(ws + OFF_WXHX),  133, 1024,  160, 1024, stream);
  launch_pack(Wxh_hy + 133*1024,1024,(bf16_t*)(ws + OFF_WXHH), 1024, 1024, 1024, 1024, stream);
  launch_pack(Whh_hy,          1024, (bf16_t*)(ws + OFF_WHH),   256, 1024,  256, 1024, stream);
  launch_pack(Wzx,              256, (bf16_t*)(ws + OFF_WZ) + 0*65536, 256, 256, 256, 256, stream);
  launch_pack(Wzh,              256, (bf16_t*)(ws + OFF_WZ) + 1*65536, 256, 256, 256, 256, stream);
  launch_pack(Wzb,              256, (bf16_t*)(ws + OFF_WZ) + 2*65536, 256, 256, 256, 256, stream);
  for (int g = 0; g < 4; ++g) {
    launch_pack(Dx + (size_t)g*64*1024, 1024, (bf16_t*)(ws + OFF_DPK) + (size_t)(0*4+g)*65536, 64, 1024, 64, 1024, stream);
    launch_pack(Dh + (size_t)g*64*1024, 1024, (bf16_t*)(ws + OFF_DPK) + (size_t)(1*4+g)*65536, 64, 1024, 64, 1024, stream);
    launch_pack(Db + (size_t)g*64*1024, 1024, (bf16_t*)(ws + OFF_DPK) + (size_t)(2*4+g)*65536, 64, 1024, 64, 1024, stream);
  }
  launch_pack(fc_prj_w,  123, (bf16_t*)(ws + OFF_PRJ),  1024, 123, 1024, 128, stream);
  launch_pack(fc_in_w,  2560, (bf16_t*)(ws + OFF_FCIN),  128, 2560, 128, 2560, stream);

  // activations -> bf16 (strokes padded 133 -> 160)
  {
    long tot = (long)NSEQ * Bb * IN_PAD;
    cvt_rowpad<<<(int)((tot + TPB - 1) / TPB), TPB, 0, stream>>>(
        strokes, (bf16_t*)(ws + OFF_XBF), (long)NSEQ * Bb, IN_DIM, IN_PAD);
    long tz = (long)Bb * LATENT;
    cvt_rowpad<<<(int)((tz + TPB - 1) / TPB), TPB, 0, stream>>>(
        z, (bf16_t*)(ws + OFF_ZBF), (long)Bb, LATENT, LATENT);
  }

  // initial state s0 = tanh(z @ fc_in_w + b)
  init_state<<<80, TPB, 0, stream>>>((const bf16_t*)(ws + OFF_ZBF),
                                     (const bf16_t*)(ws + OFF_FCIN), fc_in_b,
                                     (bf16_t*)(ws + OFF_HBF), (float*)(ws + OFF_CBUF),
                                     (bf16_t*)(ws + OFF_HHBF), (float*)(ws + OFF_CHBUF));

  // the 128-step recurrence: one persistent kernel + LDS async staging
  scan_kernel<<<NWG, TPB, LDS_BYTES, stream>>>(ws, b_hy, bzx, bzh, b0);

  // time-parallel projection + MDN head
  proj_kernel<<<512, TPB, 0, stream>>>((const bf16_t*)(ws + OFF_OUTS),
                                       (const bf16_t*)(ws + OFF_PRJ), fc_prj_b,
                                       (float*)(ws + OFF_DEC));
  mdn_kernel<<<32, TPB, 0, stream>>>((const float*)(ws + OFF_DEC), (float*)d_out);
}
